// SAGE_8186207666616
// MI455X (gfx1250) — matrix operand order repaired and checked by reading, not compiled, not run on hardware
//
#include <hip/hip_runtime.h>
#include <hip/hip_bf16.h>
#include <math.h>

// ---------------------------------------------------------------------------
// Types for WMMA
// ---------------------------------------------------------------------------
typedef __attribute__((ext_vector_type(16))) __bf16 v16bf;
typedef __attribute__((ext_vector_type(8)))  float  v8f;

union FragBF {
    v16bf v;
    unsigned int u[8];
    uint4 q[2];
};

__device__ __forceinline__ unsigned short f32_to_bf16(float f) {
    unsigned int u = __builtin_bit_cast(unsigned int, f);
    unsigned int r = u + 0x7FFFu + ((u >> 16) & 1u);   // round-to-nearest-even
    return (unsigned short)(r >> 16);
}
__device__ __forceinline__ unsigned int pack_bf16x2(float a, float b) {
    return (unsigned)f32_to_bf16(a) | ((unsigned)f32_to_bf16(b) << 16);
}

// ---------------------------------------------------------------------------
// Kernel 1: edge scatter (mean numerator + counts) via global f32 atomics.
// One wave (32 lanes) per edge; lanes stride over the feature dim (coalesced).
// ---------------------------------------------------------------------------
__global__ void sage_scatter_kernel(const float* __restrict__ feat,
                                    const int* __restrict__ src,
                                    const int* __restrict__ dst,
                                    int E, int d,
                                    float* __restrict__ agg,
                                    float* __restrict__ cnt) {
    long long idx = (long long)blockIdx.x * blockDim.x + threadIdx.x;
    int e    = (int)(idx >> 5);
    int lane = (int)(idx & 31);
    if (e >= E) return;
    int s = src[e];
    int t = dst[e];
    const float* fs = feat + (size_t)s * d;
    float*       ad = agg  + (size_t)t * d;
    for (int f = lane; f < d; f += 32) {
        atomicAdd(&ad[f], fs[f]);
    }
    if (lane == 0) atomicAdd(&cnt[t], 1.0f);
}

// ---------------------------------------------------------------------------
// Kernel 2: cnt -> 1/max(cnt,1)
// ---------------------------------------------------------------------------
__global__ void sage_invcnt_kernel(float* __restrict__ cnt, int n) {
    int i = blockIdx.x * blockDim.x + threadIdx.x;
    if (i < n) cnt[i] = 1.0f / fmaxf(cnt[i], 1.0f);
}

// ---------------------------------------------------------------------------
// Kernel 3: pack [Wl ; Wr] (f32, row-major [K0, Nact]) into bf16 WMMA B-frag
// layout: Bp[ ((s*NTpad + t)*32 + lane)*8 + i ]  (one uint = a K-pair).
// s = 32-K step over 2*KP, t = 16-col tile, lane = wave lane, i = pair index.
// Fragment rule (16-bit B 32x16): lanes 0-15: K=0..7,16..23 ; lanes 16-31: K=8..15,24..31;
// pairs sequential.
// ---------------------------------------------------------------------------
__global__ void sage_packW_kernel(const float* __restrict__ Wl,
                                  const float* __restrict__ Wr,
                                  unsigned int* __restrict__ Bp,
                                  int K0, int KP, int Nact, int NTpad) {
    int idx   = blockIdx.x * blockDim.x + threadIdx.x;
    int total = ((2 * KP) >> 5) * NTpad * 32;
    if (idx >= total) return;
    int lane = idx & 31;
    int st   = idx >> 5;               // s*NTpad + t
    int t    = st % NTpad;
    int s    = st / NTpad;
    int half = lane >> 4;
    int n    = t * 16 + (lane & 15);
    unsigned int* dst = Bp + (size_t)idx * 8;
    #pragma unroll
    for (int i = 0; i < 8; ++i) {
        int kg  = s * 32 + (i >> 2) * 16 + half * 8 + 2 * (i & 3);      // global K (over 2*KP)
        int seg = (kg >= KP) ? 1 : 0;
        int kk  = kg - seg * KP;
        const float* W = seg ? Wr : Wl;
        float v0 = 0.0f, v1 = 0.0f;
        if (n < Nact) {
            if (kk     < K0) v0 = W[(size_t)kk * Nact + n];
            if (kk + 1 < K0) v1 = W[(size_t)(kk + 1) * Nact + n];
        }
        dst[i] = pack_bf16x2(v0, v1);
    }
}

// ---------------------------------------------------------------------------
// Kernel 4: fused SAGE GEMM (WMMA bf16 -> f32):
//   out[M,Nact] = act( mean[M,K0] @ Wl + xdst[M,K0] @ Wr + b )
// mean[r,k] = agg[r,k]*inv[r] at staging time; weights pre-packed (Bp).
// Block 256 thr / 8 waves; tile 64(M) x 64(N); 64-K staged per barrier pair.
// Wave = 2 accumulators (M sub-tiles) x 2 K-sub-steps -> 4 WMMAs / iter.
// Staging is branch-free: addresses clamped, zeros injected via scale select.
// REQUIRES: K0 % 4 == 0, KP % 64 == 0.
// ---------------------------------------------------------------------------
__global__ void __launch_bounds__(256)
sage_gemm_wmma_kernel(const float* __restrict__ agg,
                      const float* __restrict__ inv,
                      const float* __restrict__ xdst,
                      const unsigned int* __restrict__ Bp,
                      const float* __restrict__ bias,
                      float* __restrict__ out,
                      int M, int Nact, int K0, int KP, int NTpad, int do_relu) {
    // A panel: 64 rows x 64 K as bf16 pairs; row stride 34 uints (pad, even
    // so 8B-aligned pair loads merge into ds_load_2addr_b64)
    __shared__ unsigned int AsU[64 * 34];
    __shared__ float invS[64];

    const int tid  = threadIdx.x;
    const int wave = tid >> 5;       // 0..7
    const int lane = tid & 31;
    const int wm   = wave >> 2;      // 0..1
    const int wn   = wave & 3;       // 0..3
    const int half = lane >> 4;
    const int l16  = lane & 15;

    const int rowBase = blockIdx.y * 64;
    const int colBase = blockIdx.x * 64;
    const int ct      = blockIdx.x * 4 + wn;   // 16-col tile index

    // cache inv[] for this block's rows
    if (tid < 64) {
        int gr = rowBase + tid;
        invS[tid] = (gr < M) ? inv[min(gr, M - 1)] : 0.0f;
    }
    __syncthreads();

    v8f acc0 = {};
    v8f acc1 = {};

    const int iters = KP >> 6;            // 64-K iterations per segment
    const int s32PerSeg = KP >> 5;        // 32-K steps per segment

    for (int seg = 0; seg < 2; ++seg) {
        const float* Asrc = seg ? xdst : agg;
        for (int it = 0; it < iters; ++it) {
            const int kk0 = it << 6;

            // ---- stage A panel: 64 rows x 16 four-K chunks, 4 per thread ----
            // branch-free: clamp addresses, zero via scale select
            #pragma unroll
            for (int c = 0; c < 4; ++c) {
                int chunk = tid + c * 256;         // 0..1023
                int r     = chunk >> 4;            // 0..63
                int c4    = chunk & 15;
                int kk    = c4 << 2;               // 0..60
                int kkl   = kk0 + kk;
                int gr    = rowBase + r;
                bool valid = (gr < M) & (kkl + 3 < K0);
                int  grc   = min(gr, M - 1);
                int  kklc  = valid ? kkl : 0;
                float4 v4 = *(const float4*)(Asrc + (size_t)grc * K0 + kklc);
                float sc = seg ? 1.0f : invS[r];
                sc = valid ? sc : 0.0f;
                unsigned int* rp = AsU + r * 34 + (kk >> 1);
                rp[0] = pack_bf16x2(v4.x * sc, v4.y * sc);
                rp[1] = pack_bf16x2(v4.z * sc, v4.w * sc);
            }
            __syncthreads();

            // ---- 2 x 32-K sub-steps ----
            #pragma unroll
            for (int j = 0; j < 2; ++j) {
                const int s32 = seg * s32PerSeg + it * 2 + j;

                // B fragment: two coalesced b128 loads from packed weights
                FragBF fb;
                {
                    const unsigned int* bp =
                        Bp + ((size_t)((s32 * NTpad + ct) * 32 + lane) * 8);
                    fb.q[0] = *(const uint4*)bp;
                    fb.q[1] = *(const uint4*)(bp + 4);
                }

                // A fragments (ISA 16-bit A 16x32 layout):
                // lanes 0-15 : pairs K={0,2,4,6},{16,18,20,22}
                // lanes16-31 : pairs K={8,10,12,14},{24,26,28,30}
                FragBF fa0, fa1;
                {
                    const int kbU = j * 16 + half * 4;
                    const unsigned int* rp0 = AsU + (wm * 16 + l16) * 34;
                    const unsigned int* rp1 = rp0 + 32 * 34;
                    #pragma unroll
                    for (int i = 0; i < 4; ++i) {
                        fa0.u[i]     = rp0[kbU + i];
                        fa0.u[4 + i] = rp0[kbU + 8 + i];
                        fa1.u[i]     = rp1[kbU + i];
                        fa1.u[4 + i] = rp1[kbU + 8 + i];
                    }
                }

                acc0 = __builtin_amdgcn_wmma_f32_16x16x32_bf16(
                           false, fa0.v, false, fb.v, (short)0, acc0, false, false);
                acc1 = __builtin_amdgcn_wmma_f32_16x16x32_bf16(
                           false, fa1.v, false, fb.v, (short)0, acc1, false, false);
            }
            __syncthreads();
        }
    }

    // ---- epilogue: bias + optional relu, masked store ----
    // C layout: lane -> col l16 ; VGPR j -> row half*8 + j
    const int colc = colBase + wn * 16 + l16;
    if (colc < Nact) {
        const float bv = bias[colc];
        #pragma unroll
        for (int j = 0; j < 8; ++j) {
            int rowc = rowBase + wm * 16 + half * 8 + j;
            if (rowc < M) {
                float v = acc0[j] + bv;
                if (do_relu) v = fmaxf(v, 0.0f);
                out[(size_t)rowc * Nact + colc] = v;
            }
            int rowc1 = rowc + 32;
            if (rowc1 < M) {
                float v = acc1[j] + bv;
                if (do_relu) v = fmaxf(v, 0.0f);
                out[(size_t)rowc1 * Nact + colc] = v;
            }
        }
    }
}

// ---------------------------------------------------------------------------
// Kernel 5: row-wise log_softmax, one wave per row (cols <= 64)
// ---------------------------------------------------------------------------
__global__ void sage_logsoftmax_kernel(const float* __restrict__ logits,
                                       float* __restrict__ out,
                                       int rows, int cols) {
    int gwave = (int)(((long long)blockIdx.x * blockDim.x + threadIdx.x) >> 5);
    int lane  = threadIdx.x & 31;
    if (gwave >= rows) return;
    const float* xr = logits + (size_t)gwave * cols;
    float v0 = (lane      < cols) ? xr[lane]      : -INFINITY;
    float v1 = (lane + 32 < cols) ? xr[lane + 32] : -INFINITY;
    float m = fmaxf(v0, v1);
    #pragma unroll
    for (int o = 16; o > 0; o >>= 1) m = fmaxf(m, __shfl_xor(m, o, 32));
    float s = ((lane      < cols) ? expf(v0 - m) : 0.0f) +
              ((lane + 32 < cols) ? expf(v1 - m) : 0.0f);
    #pragma unroll
    for (int o = 16; o > 0; o >>= 1) s += __shfl_xor(s, o, 32);
    float ls = logf(s);
    if (lane      < cols) out[(size_t)gwave * cols + lane]      = v0 - m - ls;
    if (lane + 32 < cols) out[(size_t)gwave * cols + lane + 32] = v1 - m - ls;
}

// ---------------------------------------------------------------------------
// Host-side orchestration
// ---------------------------------------------------------------------------
extern "C" void kernel_launch(void* const* d_in, const int* in_sizes, int n_in,
                              void* d_out, int out_size, void* d_ws, size_t ws_size,
                              hipStream_t stream) {
    (void)n_in; (void)out_size; (void)ws_size;

    const int N1 = 200000, N2 = 20000, B = 4096;
    const int DIN = 100, DH = 256, DOUT = 47;

    const float* x    = (const float*)d_in[0];
    const int*   src0 = (const int*)d_in[1];
    const int*   dst0 = (const int*)d_in[2];
    const int*   src1 = (const int*)d_in[3];
    const int*   dst1 = (const int*)d_in[4];
    const int*   src2 = (const int*)d_in[5];
    const int*   dst2 = (const int*)d_in[6];
    const float* Wl0  = (const float*)d_in[7];
    const float* Wr0  = (const float*)d_in[8];
    const float* b0   = (const float*)d_in[9];
    const float* Wl1  = (const float*)d_in[10];
    const float* Wr1  = (const float*)d_in[11];
    const float* b1   = (const float*)d_in[12];
    const float* Wl2  = (const float*)d_in[13];
    const float* Wr2  = (const float*)d_in[14];
    const float* b2   = (const float*)d_in[15];

    const int E0 = in_sizes[1];
    const int E1 = in_sizes[3];
    const int E2 = in_sizes[5];

    // layer GEMM geometry
    const int KP0 = 128, KP12 = 256;
    const int NT0 = 16;                 // Nact=256 -> 4 col blocks * 4 tiles
    const int NT1 = 16;
    const int NT2 = 4;                  // Nact=47  -> 1 col block * 4 tiles
    const size_t BP0 = (size_t)((2 * KP0)  >> 5) * NT0 * 32 * 8;  // uints
    const size_t BP1 = (size_t)((2 * KP12) >> 5) * NT1 * 32 * 8;
    const size_t BP2 = (size_t)((2 * KP12) >> 5) * NT2 * 32 * 8;

    // Workspace layout (float-sized slots, 256B-aligned chunks)
    float* ws = (float*)d_ws;
    size_t off = 0;
    auto alloc = [&](size_t nfloats) -> float* {
        float* p = ws + off;
        off += (nfloats + 63) & ~(size_t)63;
        return p;
    };
    float* agg0   = alloc((size_t)N1 * DIN);
    float* cnt0   = alloc(N1);
    float* h0     = alloc((size_t)N1 * DH);
    float* agg1   = alloc((size_t)N2 * DH);
    float* cnt1   = alloc(N2);
    float* h1     = alloc((size_t)N2 * DH);
    float* agg2   = alloc((size_t)B * DH);
    float* cnt2   = alloc(B);
    float* logits = alloc((size_t)B * DOUT);
    unsigned int* Bp0 = (unsigned int*)alloc(BP0);
    unsigned int* Bp1 = (unsigned int*)alloc(BP1);
    unsigned int* Bp2 = (unsigned int*)alloc(BP2);

    const dim3 blk(256);

    // ---- pack all weights up front (tiny) ----
    {
        int t0 = ((2 * KP0)  >> 5) * NT0 * 32;
        int t1 = ((2 * KP12) >> 5) * NT1 * 32;
        int t2 = ((2 * KP12) >> 5) * NT2 * 32;
        sage_packW_kernel<<<(t0 + 255) / 256, blk, 0, stream>>>(Wl0, Wr0, Bp0, DIN, KP0, DH, NT0);
        sage_packW_kernel<<<(t1 + 255) / 256, blk, 0, stream>>>(Wl1, Wr1, Bp1, DH, KP12, DH, NT1);
        sage_packW_kernel<<<(t2 + 255) / 256, blk, 0, stream>>>(Wl2, Wr2, Bp2, DH, KP12, DOUT, NT2);
    }

    // ---------------- Layer 0 ----------------
    hipMemsetAsync(agg0, 0, (size_t)N1 * DIN * sizeof(float), stream);
    hipMemsetAsync(cnt0, 0, (size_t)N1 * sizeof(float), stream);
    {
        long long th = (long long)E0 * 32;
        sage_scatter_kernel<<<(unsigned)((th + 255) / 256), blk, 0, stream>>>(
            x, src0, dst0, E0, DIN, agg0, cnt0);
        sage_invcnt_kernel<<<(N1 + 255) / 256, blk, 0, stream>>>(cnt0, N1);
        dim3 grid((DH + 63) / 64, (N1 + 63) / 64);
        sage_gemm_wmma_kernel<<<grid, blk, 0, stream>>>(
            agg0, cnt0, x, Bp0, b0, h0, N1, DH, DIN, KP0, NT0, 1);
    }

    // ---------------- Layer 1 ----------------
    hipMemsetAsync(agg1, 0, (size_t)N2 * DH * sizeof(float), stream);
    hipMemsetAsync(cnt1, 0, (size_t)N2 * sizeof(float), stream);
    {
        long long th = (long long)E1 * 32;
        sage_scatter_kernel<<<(unsigned)((th + 255) / 256), blk, 0, stream>>>(
            h0, src1, dst1, E1, DH, agg1, cnt1);
        sage_invcnt_kernel<<<(N2 + 255) / 256, blk, 0, stream>>>(cnt1, N2);
        dim3 grid((DH + 63) / 64, (N2 + 63) / 64);
        sage_gemm_wmma_kernel<<<grid, blk, 0, stream>>>(
            agg1, cnt1, h0, Bp1, b1, h1, N2, DH, DH, KP12, NT1, 1);
    }

    // ---------------- Layer 2 ----------------
    hipMemsetAsync(agg2, 0, (size_t)B * DH * sizeof(float), stream);
    hipMemsetAsync(cnt2, 0, (size_t)B * sizeof(float), stream);
    {
        long long th = (long long)E2 * 32;
        sage_scatter_kernel<<<(unsigned)((th + 255) / 256), blk, 0, stream>>>(
            h1, src2, dst2, E2, DH, agg2, cnt2);
        sage_invcnt_kernel<<<(B + 255) / 256, blk, 0, stream>>>(cnt2, B);
        dim3 grid((DOUT + 63) / 64, (B + 63) / 64);
        sage_gemm_wmma_kernel<<<grid, blk, 0, stream>>>(
            agg2, cnt2, h1, Bp2, b2, logits, B, DOUT, DH, KP12, NT2, 0);
    }

    // ---------------- log-softmax ----------------
    {
        long long th = (long long)B * 32;
        sage_logsoftmax_kernel<<<(unsigned)((th + 255) / 256), blk, 0, stream>>>(
            logits, (float*)d_out, B, DOUT);
    }
}
